// NLSRUpsampler_58755152609721
// MI455X (gfx1250) — matrix operand order repaired and checked by reading, not compiled, not run on hardware
//
#include <hip/hip_runtime.h>
#include <hip/hip_bf16.h>

typedef __attribute__((ext_vector_type(16))) _Float16 v16h;
typedef __attribute__((ext_vector_type(8)))  _Float16 v8h;
typedef __attribute__((ext_vector_type(8)))  float    v8f;
typedef __attribute__((address_space(3)))    _Float16 lds_f16;

#define B_     4
#define L_     9216          // 96*96 tokens after upsample
#define HH     96
#define WW     96
#define CM_    32            // match channels
#define CO_    64
#define NH_    4
#define LF_    (NH_*L_)      // 36864 flat sorted length per batch
#define CHUNK_ 144
#define NKC_   64            // chunks per hash round (L/CHUNK)
#define NB_    256           // total code buckets = NH*64

__device__ __forceinline__ int clampi(int v, int lo, int hi) {
  return v < lo ? lo : (v > hi ? hi : v);
}

// ---- fragment loaders (contiguous k = 16h+s order, same in A and B), K-contiguous memory ----
// A (16x32 f16): lane holds K = {off..off+15}, off = (lane>=16)?16:0
__device__ __forceinline__ v16h loadA32(const _Float16* rowp, int lane) {
  const int off = (lane & 16) ? 16 : 0;
  v8h lo = *(const v8h*)(rowp + off);
  v8h hi = *(const v8h*)(rowp + off + 8);
  v16h f;
#pragma unroll
  for (int e = 0; e < 8; ++e) { f[e] = lo[e]; f[e + 8] = hi[e]; }
  return f;
}
// B (32x16 f16): lane holds 16 contiguous K starting at caller-resolved base
__device__ __forceinline__ v16h load16(const _Float16* p) {
  v8h lo = *(const v8h*)p;
  v8h hi = *(const v8h*)(p + 8);
  v16h f;
#pragma unroll
  for (int e = 0; e < 8; ++e) { f[e] = lo[e]; f[e + 8] = hi[e]; }
  return f;
}
__device__ __forceinline__ v16h zero16() {
  v16h f;
#pragma unroll
  for (int e = 0; e < 16; ++e) f[e] = (_Float16)0.f;
  return f;
}

// ---------------- bicubic upsample (Keys a=-0.5, half-pixel) -> NLC f16 ----------------
__device__ __forceinline__ float cubw(float t) {
  t = fabsf(t);
  if (t <= 1.f) return (1.5f * t - 2.5f) * t * t + 1.f;
  if (t < 2.f)  return ((-0.5f * t + 2.5f) * t - 4.f) * t + 2.f;
  return 0.f;
}

__global__ void k_upsample(const float* __restrict__ x, _Float16* __restrict__ uph) {
  int id = blockIdx.x * blockDim.x + threadIdx.x;
  if (id >= B_ * 64 * L_) return;
  int c = id % 64;  int t = (id / 64) % L_;  int b = id / (64 * L_);
  int yo = t / WW, xo = t % WW;
  float py = (yo + 0.5f) * 0.5f - 0.5f;
  float px = (xo + 0.5f) * 0.5f - 0.5f;
  int iy = (int)floorf(py), ix = (int)floorf(px);
  const float* src = x + ((size_t)(b * 64 + c)) * 48 * 48;
  float s = 0.f;
#pragma unroll
  for (int dy = -1; dy <= 2; ++dy) {
    float wy = cubw(py - (float)(iy + dy));
    int yy = clampi(iy + dy, 0, 47);
#pragma unroll
    for (int dx = -1; dx <= 2; ++dx) {
      float wx = cubw(px - (float)(ix + dx));
      int xx = clampi(ix + dx, 0, 47);
      s += wy * wx * src[yy * 48 + xx];
    }
  }
  uph[id] = (_Float16)s;   // [b][t][c]
}

// ---------------- weight pack: OIHW f32 -> [co][r*CI+ci] f16 (r-major, ci-minor) ----------------
__global__ void k_packw(const float* __restrict__ s, _Float16* __restrict__ d,
                        int CO, int CI, int KSS) {
  int i = blockIdx.x * blockDim.x + threadIdx.x;
  int KT = CI * KSS;
  if (i >= CO * KT) return;
  int co = i / KT, k = i % KT;
  int r = k / CI, ci = k % CI;
  d[i] = (_Float16)s[(size_t)(co * CI + ci) * KSS + r];
}

// rot [f][h*32+i] f32 -> rotT [n=h*32+i][f] f16
__global__ void k_packrot(const float* __restrict__ s, _Float16* __restrict__ d) {
  int i = blockIdx.x * blockDim.x + threadIdx.x;
  if (i >= 128 * 32) return;
  int n = i / 32, f = i % 32;
  d[i] = (_Float16)s[f * 128 + n];
}

// ---------------- implicit-im2col conv as WMMA GEMM (NLC f16 in/out) ----------------
template <int CI, int CO, int KS, bool RELU>
__global__ void __launch_bounds__(32)
k_conv_wmma(const _Float16* __restrict__ in,   // [B][L][CI]
            const _Float16* __restrict__ wh,   // [CO][KS*KS*CI]
            const float* __restrict__ bias,
            _Float16* __restrict__ outh) {     // [B][L][CO]
  const int lane = threadIdx.x;
  const int p0  = blockIdx.x * 16;
  const int co0 = blockIdx.y * 16;
  const int b   = blockIdx.z;
  const int KT  = CI * KS * KS;
  const int pn  = p0 + (lane & 15);
  const int py  = pn / WW, px = pn % WW;
  const _Float16* wrow = wh + (size_t)(co0 + (lane & 15)) * KT;
  const int koffB = (lane & 16) ? 16 : 0;
  v8f acc = {0.f, 0.f, 0.f, 0.f, 0.f, 0.f, 0.f, 0.f};
  for (int k0 = 0; k0 < KT; k0 += 32) {
    if (k0 + 32 < KT) __builtin_prefetch(wrow + k0 + 32, 0, 3);
    v16h av = loadA32(wrow + k0, lane);
    int kb = k0 + koffB;                 // 16 contiguous K, single (kh,kw) tap
    int r  = kb / CI, ci0 = kb % CI;
    int y  = py + r / KS - (KS / 2);
    int x  = px + r % KS - (KS / 2);
    v16h bv;
    if (y >= 0 && y < HH && x >= 0 && x < WW)
      bv = load16(in + ((size_t)(b) * L_ + y * WW + x) * CI + ci0);
    else
      bv = zero16();
    acc = __builtin_amdgcn_wmma_f32_16x16x32_f16(false, av, false, bv,
                                                 (short)0, acc, false, false);
  }
  v8h o;
#pragma unroll
  for (int r = 0; r < 8; ++r) {
    int m = r + ((lane & 16) ? 8 : 0);
    float v = acc[r] + bias[co0 + m];
    if (RELU) v = fmaxf(v, 0.f);
    o[r] = (_Float16)v;
  }
  // contiguous 8 couts per lane -> one b128 store
  *(v8h*)(outh + ((size_t)b * L_ + pn) * CO + co0 + ((lane & 16) ? 8 : 0)) = o;
}

// ---------------- LSH rotation projection: rv[b][t][h*32+i] ----------------
__global__ void __launch_bounds__(32)
k_rot_wmma(const _Float16* __restrict__ xeh, const _Float16* __restrict__ rotT,
           float* __restrict__ rv) {
  const int lane = threadIdx.x;
  const int t0 = blockIdx.x * 16, n0 = blockIdx.y * 16, b = blockIdx.z;
  v16h av = loadA32(xeh + ((size_t)b * L_ + t0 + (lane & 15)) * CM_, lane);
  v16h bv = load16(rotT + (size_t)(n0 + (lane & 15)) * CM_ + ((lane & 16) ? 16 : 0));
  v8f acc = {0.f, 0.f, 0.f, 0.f, 0.f, 0.f, 0.f, 0.f};
  acc = __builtin_amdgcn_wmma_f32_16x16x32_f16(false, av, false, bv,
                                               (short)0, acc, false, false);
#pragma unroll
  for (int r = 0; r < 8; ++r) {
    int m = r + ((lane & 16) ? 8 : 0);
    rv[((size_t)b * L_ + t0 + m) * 128 + n0 + (lane & 15)] = acc[r];
  }
}

// ---------------- argmax over [rv, -rv] -> bucket codes ----------------
__global__ void k_codes(const float* __restrict__ rv, int* __restrict__ codes) {
  int id = blockIdx.x * blockDim.x + threadIdx.x;
  if (id >= B_ * NH_ * L_) return;
  int t = id % L_;
  int h = (id / L_) % NH_;
  int b = id / (L_ * NH_);
  const float* p = rv + ((size_t)b * L_ + t) * 128 + h * 32;
  float best = -3.4e38f; int bi = 0;
  for (int j = 0; j < 64; ++j) {
    float v = (j < 32) ? p[j] : -p[j - 32];
    if (v > best) { best = v; bi = j; }
  }
  codes[(size_t)b * LF_ + h * L_ + t] = bi + h * 64;
}

// ---------------- counting-sort: LDS histogram + exclusive scan ----------------
__global__ void k_hist(const int* __restrict__ codes, int* __restrict__ binst) {
  __shared__ int hst[NB_];
  int b = blockIdx.x, tid = threadIdx.x;
  hst[tid] = 0;
  __syncthreads();
  const int* c = codes + (size_t)b * LF_;
  for (int i = tid; i < LF_; i += 256) atomicAdd(&hst[c[i]], 1);
  __syncthreads();
  if (tid == 0) {
    int s = 0;
    for (int k = 0; k < NB_; ++k) { int v = hst[k]; binst[b * NB_ + k] = s; s += v; }
  }
}

// stable scatter: thread t owns code value t, scans in original order
__global__ void k_scatter(const int* __restrict__ codes, const int* __restrict__ binst,
                          int* __restrict__ sidx) {
  int b = blockIdx.x, cc = threadIdx.x;
  int off = binst[b * NB_ + cc];
  const int* c = codes + (size_t)b * LF_;
  int* si = sidx + (size_t)b * LF_;
  for (int i = 0; i < LF_; ++i)
    if (c[i] == cc) si[off++] = i;
}

__global__ void k_undo(const int* __restrict__ sidx, int* __restrict__ undo) {
  int id = blockIdx.x * blockDim.x + threadIdx.x;
  if (id >= B_ * LF_) return;
  int b = id / LF_, j = id % LF_;
  undo[(size_t)b * LF_ + sidx[id]] = j;
}

// ---------------- gather sorted tokens + F.normalize(eps=5e-5); ys transposed ----------------
__global__ void k_gather(const int* __restrict__ sidx,
                         const _Float16* __restrict__ xeh, const _Float16* __restrict__ yeh,
                         _Float16* __restrict__ xsh, _Float16* __restrict__ xmnh,
                         _Float16* __restrict__ ysT) {
  int id = blockIdx.x * blockDim.x + threadIdx.x;
  if (id >= B_ * LF_) return;
  int b = id / LF_, j = id % LF_;
  int src = sidx[id] % L_;
  const _Float16* xe = xeh + ((size_t)b * L_ + src) * CM_;
  float tv[CM_];
  float nrm = 0.f;
#pragma unroll
  for (int c = 0; c < CM_; ++c) { float v = (float)xe[c]; tv[c] = v; nrm += v * v; }
  nrm = fmaxf(sqrtf(nrm), 5e-5f);
  _Float16* xo = xsh  + (size_t)id * CM_;
  _Float16* xn = xmnh + (size_t)id * CM_;
#pragma unroll
  for (int c = 0; c < CM_; ++c) { xo[c] = (_Float16)tv[c]; xn[c] = (_Float16)(tv[c] / nrm); }
  const _Float16* ye = yeh + ((size_t)b * L_ + src) * CO_;
#pragma unroll
  for (int c = 0; c < CO_; ++c)
    ysT[((size_t)(b * CO_ + c)) * LF_ + j] = ye[c];     // [b][co][flat j]
}

// ---------------- chunked attention: raw=xs@xm^T, lse, ret=exp(raw-lse)@y3 ----------------
// Key tile (432x32 f16) is staged to LDS with gfx1250 async-to-LDS DMA; the
// score tile aliases the same LDS pool after the key tile is consumed.
__global__ void __launch_bounds__(32)
k_attn(const _Float16* __restrict__ xsh, const _Float16* __restrict__ xmnh,
       const _Float16* __restrict__ ysT, float* __restrict__ ret,
       float* __restrict__ bsc) {
  __shared__ _Float16 pool[432 * 32];  // 27.6KB: key tile, later aliased by score tile
  __shared__ float    raw[16][433];    // 16 query rows x 432 keys (padded)
  _Float16* xkey = pool;                                // [key][32]
  _Float16 (*scx)[448] = (_Float16(*)[448])pool;        // 16 x 448 halfs (14KB)

  const int lane = threadIdx.x;
  const int m0 = blockIdx.x * 16;
  const int kc = blockIdx.y;
  const int h  = blockIdx.z & 3, b = blockIdx.z >> 2;
  const size_t base = (size_t)b * LF_;
  const int jrow = h * L_ + kc * CHUNK_ + m0;            // sorted flat row (per batch)
  const int kcb = (kc + NKC_ - 1) % NKC_, kcf = (kc + 1) % NKC_;
  const int koffB = (lane & 16) ? 16 : 0;

  // ---- async DMA: stage 432 key rows (64B each) into LDS, 16B per lane ----
  unsigned lbase = (unsigned)(unsigned long long)(lds_f16*)pool;
  for (int it = 0; it < 54; ++it) {                      // 54*32 lanes = 1728 x16B = 27648B
    int chunk = it * 32 + lane;
    int row = chunk >> 2;                                // key index 0..431
    int cs = row / CHUNK_, r = row % CHUNK_;
    int kc2 = (cs == 0) ? kc : (cs == 1 ? kcb : kcf);
    const _Float16* gsrc =
        xmnh + (base + h * L_ + (size_t)kc2 * CHUNK_ + r) * CM_ + (chunk & 3) * 8;
    unsigned ldst = lbase + (unsigned)chunk * 16;
    asm volatile("global_load_async_to_lds_b128 %0, %1, off"
                 :: "v"(ldst), "v"(gsrc) : "memory");
  }
  asm volatile("s_wait_asynccnt 0x0" ::: "memory");
  __syncthreads();

  v16h av = loadA32(xsh + (base + jrow + (lane & 15)) * CM_, lane);

  for (int nt = 0; nt < 27; ++nt) {                      // 432 keys / 16
    int col = nt * 16 + (lane & 15);
    v16h bv = load16(xkey + (size_t)col * CM_ + koffB);  // ds_load_b128 pair
    v8f c0 = {0.f, 0.f, 0.f, 0.f, 0.f, 0.f, 0.f, 0.f};
    c0 = __builtin_amdgcn_wmma_f32_16x16x32_f16(false, av, false, bv,
                                                (short)0, c0, false, false);
#pragma unroll
    for (int r2 = 0; r2 < 8; ++r2) {
      int m = r2 + ((lane & 16) ? 8 : 0);
      raw[m][col] = c0[r2];
    }
  }
  __syncthreads();                                       // key tile dead beyond here
  if (lane < 16) {
    float mx = -3.4e38f;
    for (int cI = 0; cI < 432; ++cI) mx = fmaxf(mx, raw[lane][cI]);
    float s = 0.f;
    for (int cI = 0; cI < 432; ++cI) s += expf(raw[lane][cI] - mx);
    float lse = mx + logf(s);
    for (int cI = 0; cI < 432; ++cI) scx[lane][cI] = (_Float16)expf(raw[lane][cI] - lse);
    for (int cI = 432; cI < 448; ++cI) scx[lane][cI] = (_Float16)0.f;
    bsc[base + jrow + lane] = lse;
  }
  __syncthreads();
  for (int nt = 0; nt < 4; ++nt) {                       // 64 out channels / 16
    const _Float16* ycol = ysT + ((size_t)(b * CO_ + nt * 16 + (lane & 15))) * LF_;
    __builtin_prefetch(ycol + h * L_ + kc * CHUNK_, 0, 3);
    v8f acc = {0.f, 0.f, 0.f, 0.f, 0.f, 0.f, 0.f, 0.f};
    for (int k0 = 0; k0 < 448; k0 += 32) {
      // A from LDS score (ds_load_b128 pairs)
      v16h a2 = loadA32(&scx[lane & 15][k0], lane);
      int kk0 = k0 + koffB;                              // 16-key run, never crosses chunks
      v16h b2;
      if (kk0 < 432) {
        int cs = kk0 / CHUNK_, r0 = kk0 % CHUNK_;
        int kc2 = (cs == 0) ? kc : (cs == 1 ? kcb : kcf);
        b2 = load16(ycol + h * L_ + (size_t)kc2 * CHUNK_ + r0);
      } else {
        b2 = zero16();
      }
      acc = __builtin_amdgcn_wmma_f32_16x16x32_f16(false, a2, false, b2,
                                                   (short)0, acc, false, false);
    }
#pragma unroll
    for (int r2 = 0; r2 < 8; ++r2) {
      int m = r2 + ((lane & 16) ? 8 : 0);
      ret[(base + jrow + m) * CO_ + nt * 16 + (lane & 15)] = acc[r2];
    }
  }
}

// ---------------- unsort, softmax over hash rounds, residual ----------------
__global__ void k_combine(const float* __restrict__ ret, const float* __restrict__ bsc,
                          const int* __restrict__ undo, const _Float16* __restrict__ a2h,
                          float* __restrict__ out) {
  int id = blockIdx.x * blockDim.x + threadIdx.x;
  if (id >= B_ * L_) return;
  int b = id / L_, t = id % L_;
  int u[NH_]; float ls[NH_];
#pragma unroll
  for (int h = 0; h < NH_; ++h) {
    u[h]  = undo[(size_t)b * LF_ + h * L_ + t];
    ls[h] = bsc[(size_t)b * LF_ + u[h]];
  }
  float mx = fmaxf(fmaxf(ls[0], ls[1]), fmaxf(ls[2], ls[3]));
  float p[NH_]; float s = 0.f;
#pragma unroll
  for (int h = 0; h < NH_; ++h) { p[h] = expf(ls[h] - mx); s += p[h]; }
  float inv = 1.f / s;
  const _Float16* resid = a2h + (size_t)id * CO_;
  for (int co = 0; co < CO_; ++co) {
    float acc = 0.f;
#pragma unroll
    for (int h = 0; h < NH_; ++h)
      acc += p[h] * ret[((size_t)b * LF_ + u[h]) * CO_ + co];
    out[((size_t)(b * CO_) + co) * L_ + t] = acc * inv + (float)resid[co];
  }
}

extern "C" void kernel_launch(void* const* d_in, const int* in_sizes, int n_in,
                              void* d_out, int out_size, void* d_ws, size_t ws_size,
                              hipStream_t stream) {
  const float* x   = (const float*)d_in[0];
  const float* w1  = (const float*)d_in[1];
  const float* b1  = (const float*)d_in[2];
  const float* w2  = (const float*)d_in[3];
  const float* b2  = (const float*)d_in[4];
  const float* wm  = (const float*)d_in[5];
  const float* bm  = (const float*)d_in[6];
  const float* wa  = (const float*)d_in[7];
  const float* ba  = (const float*)d_in[8];
  const float* rot = (const float*)d_in[9];
  float* out = (float*)d_out;

  char* w = (char*)d_ws;
  size_t off = 0;
  auto alloc = [&](size_t bytes) -> void* {
    void* p = w + off;
    off = (off + bytes + 255) & ~(size_t)255;
    return p;
  };
  _Float16* uph  = (_Float16*)alloc(sizeof(_Float16) * B_ * L_ * 64);
  _Float16* a1h  = (_Float16*)alloc(sizeof(_Float16) * B_ * L_ * 64);
  _Float16* a2h  = (_Float16*)alloc(sizeof(_Float16) * B_ * L_ * 64);
  _Float16* w1p  = (_Float16*)alloc(sizeof(_Float16) * 64 * 576);
  _Float16* w2p  = (_Float16*)alloc(sizeof(_Float16) * 64 * 576);
  _Float16* wmp  = (_Float16*)alloc(sizeof(_Float16) * 32 * 576);
  _Float16* wap  = (_Float16*)alloc(sizeof(_Float16) * 64 * 64);
  _Float16* rotT = (_Float16*)alloc(sizeof(_Float16) * 128 * 32);
  _Float16* xeh  = (_Float16*)alloc(sizeof(_Float16) * B_ * L_ * CM_);
  _Float16* yeh  = (_Float16*)alloc(sizeof(_Float16) * B_ * L_ * CO_);
  float*    rv   = (float*)   alloc(sizeof(float) * B_ * L_ * 128);
  int*      cod  = (int*)     alloc(sizeof(int) * B_ * LF_);
  int*      bst  = (int*)     alloc(sizeof(int) * B_ * NB_);
  int*      sidx = (int*)     alloc(sizeof(int) * B_ * LF_);
  int*      und  = (int*)     alloc(sizeof(int) * B_ * LF_);
  _Float16* xsh  = (_Float16*)alloc(sizeof(_Float16) * B_ * LF_ * CM_);
  _Float16* xmnh = (_Float16*)alloc(sizeof(_Float16) * B_ * LF_ * CM_);
  _Float16* ysT  = (_Float16*)alloc(sizeof(_Float16) * B_ * CO_ * LF_);
  float*    retb = (float*)   alloc(sizeof(float) * B_ * LF_ * CO_);
  float*    bsc  = (float*)   alloc(sizeof(float) * B_ * LF_);
  (void)ws_size; (void)in_sizes; (void)n_in; (void)out_size;

  int tot = B_ * 64 * L_;
  k_upsample<<<(tot + 255) / 256, 256, 0, stream>>>(x, uph);

  k_packw<<<(64 * 576 + 255) / 256, 256, 0, stream>>>(w1, w1p, 64, 64, 9);
  k_packw<<<(64 * 576 + 255) / 256, 256, 0, stream>>>(w2, w2p, 64, 64, 9);
  k_packw<<<(32 * 576 + 255) / 256, 256, 0, stream>>>(wm, wmp, 32, 64, 9);
  k_packw<<<(64 * 64 + 255) / 256, 256, 0, stream>>>(wa, wap, 64, 64, 1);
  k_packrot<<<(128 * 32 + 255) / 256, 256, 0, stream>>>(rot, rotT);

  k_conv_wmma<64, 64, 3, true ><<<dim3(L_ / 16, 4, B_), 32, 0, stream>>>(uph, w1p, b1, a1h);
  k_conv_wmma<64, 64, 3, true ><<<dim3(L_ / 16, 4, B_), 32, 0, stream>>>(a1h, w2p, b2, a2h);
  k_conv_wmma<64, 32, 3, false><<<dim3(L_ / 16, 2, B_), 32, 0, stream>>>(a2h, wmp, bm, xeh);
  k_conv_wmma<64, 64, 1, false><<<dim3(L_ / 16, 4, B_), 32, 0, stream>>>(a2h, wap, ba, yeh);

  k_rot_wmma<<<dim3(L_ / 16, 8, B_), 32, 0, stream>>>(xeh, rotT, rv);
  k_codes<<<(B_ * NH_ * L_ + 255) / 256, 256, 0, stream>>>(rv, cod);

  k_hist<<<B_, 256, 0, stream>>>(cod, bst);
  k_scatter<<<B_, 256, 0, stream>>>(cod, bst, sidx);
  k_undo<<<(B_ * LF_ + 255) / 256, 256, 0, stream>>>(sidx, und);
  k_gather<<<(B_ * LF_ + 255) / 256, 256, 0, stream>>>(sidx, xeh, yeh, xsh, xmnh, ysT);

  k_attn<<<dim3(CHUNK_ / 16, NKC_, B_ * NH_), 32, 0, stream>>>(xsh, xmnh, ysT, retb, bsc);
  k_combine<<<(B_ * L_ + 255) / 256, 256, 0, stream>>>(retb, bsc, und, a2h, out);
}
